// ESNReservoir_20315195310235
// MI455X (gfx1250) — matrix operand order repaired and checked by reading, not compiled, not run on hardware
//
#include <hip/hip_runtime.h>
#include <math.h>

// ---- types ----
typedef __bf16 v16bf __attribute__((ext_vector_type(16)));
typedef float  v8f   __attribute__((ext_vector_type(8)));
typedef float  v4f   __attribute__((ext_vector_type(4)));

// tanh: prefer single TRANS instruction v_tanh_f32 on gfx1250
#if defined(__has_builtin)
#  if __has_builtin(__builtin_amdgcn_tanhf)
#    define ESN_TANH(x) __builtin_amdgcn_tanhf(x)
#  endif
#endif
#ifndef ESN_TANH
#  define ESN_TANH(x) tanhf(x)
#endif

// exchange value with lane^16 (half-wave swap), single ds_swizzle (SWAPX16)
__device__ __forceinline__ float swap_half(float x) {
    int i = __float_as_int(x);
    i = __builtin_amdgcn_ds_swizzle(i, 0x401f); // xor=0x10, or=0, and=0x1f
    return __int_as_float(i);
}

// D = A*B + C  (bf16 in, f32 accumulate), 16x16x32
__device__ __forceinline__ v8f wmma_bf16(v16bf a, v16bf b, v8f c) {
    return __builtin_amdgcn_wmma_f32_16x16x32_bf16(
        /*neg_a=*/false, a, /*neg_b=*/false, b,
        /*c_mod=*/(short)0, c, /*reuse_a=*/false, /*reuse_b=*/false);
}

// One wave (32 threads) owns one strip of 16 batch rows and iterates all T
// steps with weights resident in VGPRs.  Transposed recurrence:
//   Y^T[u][b] = tanh( sum_j W_rec[j][u] * Y^T[j][b] + sum_d W_in[d][u]*x[b][d] + bias[u] )
// A-operand = [W_rec^T | W_in^T] (constant), B-operand = Y^T (rebuilt in-register).
__global__ __launch_bounds__(32, 1)
void esn_reservoir_kernel(const float* __restrict__ inp,   // [B, T, 8]
                          const float* __restrict__ W_in,  // [8, 64]
                          const float* __restrict__ bias,  // [64]
                          const float* __restrict__ W_rec, // [64, 64]
                          float* __restrict__ out,         // [B, T, 64]
                          int T)
{
    const int lane  = threadIdx.x;       // 0..31 (wave32)
    const int m     = lane & 15;         // row-in-tile / batch column
    const int h     = lane >> 4;         // half-wave index
    const int bstrip = blockIdx.x * 16;  // first batch row of this strip

    // ---------------- constant A fragments (loaded once) ----------------
    // A k order chosen here (same as B): elem e -> K = base + h*16 + e
    v16bf Arec[4][2];                    // W_rec^T : A[u][j] = W_rec[j][u]
    #pragma unroll
    for (int tt = 0; tt < 4; ++tt) {
        const int uM = tt * 16 + m;
        #pragma unroll
        for (int f = 0; f < 2; ++f) {
            #pragma unroll
            for (int e = 0; e < 16; ++e) {
                int K = f * 32 + h * 16 + e;
                Arec[tt][f][e] = (__bf16)W_rec[K * 64 + uM];
            }
        }
    }
    v16bf Ain[4];                        // W_in^T : A[u][d] = W_in[d][u], d<8
    #pragma unroll
    for (int tt = 0; tt < 4; ++tt) {
        const int uM = tt * 16 + m;
        #pragma unroll
        for (int e = 0; e < 16; ++e) {
            int K = h * 16 + e;
            Ain[tt][e] = (K < 8) ? (__bf16)W_in[K * 64 + uM] : (__bf16)0.0f;
        }
    }
    v8f Cinit[4];                        // bias broadcast in C/D layout
    #pragma unroll
    for (int tt = 0; tt < 4; ++tt) {
        #pragma unroll
        for (int r = 0; r < 8; ++r)
            Cinit[tt][r] = bias[tt * 16 + r + 8 * h];
    }

    // ---------------- state B fragments (Y^T), init = 0 ----------------
    v16bf Bst[2];
    #pragma unroll
    for (int f = 0; f < 2; ++f)
        #pragma unroll
        for (int e = 0; e < 16; ++e)
            Bst[f][e] = (__bf16)0.0f;

    // ---------------- streaming pointers ----------------
    const float* inBase  = inp + (size_t)(bstrip + m) * (size_t)T * 8;
    float*       outBase = out + (size_t)(bstrip + m) * (size_t)T * 64;

    // prefetch inputs for t = 0 (lanes 16..31 load same addr as lane-16; harmless)
    v4f curA = *(const v4f*)(inBase + 0);
    v4f curB = *(const v4f*)(inBase + 4);

    for (int t = 0; t < T; ++t) {
        // ---- prefetch next step's inputs early (off the critical chain) ----
        v4f nxtA = curA, nxtB = curB;
        if (t + 1 < T) {
            nxtA = *(const v4f*)(inBase + (size_t)(t + 1) * 8);
            nxtB = *(const v4f*)(inBase + (size_t)(t + 1) * 8 + 4);
        }

        // ---- input B fragment: B[K][n], K = h*16 + e; nonzero only h==0, e<8 ----
        v16bf Bin;
        #pragma unroll
        for (int e = 0; e < 16; ++e) {
            float v = 0.0f;
            if (e < 4)      v = curA[e];
            else if (e < 8) v = curB[e - 4];
            Bin[e] = (h == 0) ? (__bf16)v : (__bf16)0.0f;
        }

        // ---- 4 M-tiles x 3 K-fragments of WMMA ----
        v8f acc[4];
        #pragma unroll
        for (int tt = 0; tt < 4; ++tt) {
            v8f c = Cinit[tt];
            c = wmma_bf16(Arec[tt][0], Bst[0], c);
            c = wmma_bf16(Arec[tt][1], Bst[1], c);
            c = wmma_bf16(Ain[tt],     Bin,    c);
            acc[tt] = c;
        }

        // ---- tanh + streaming NT store: lane holds u = tt*16 + 8h + r, batch = m ----
        #pragma unroll
        for (int tt = 0; tt < 4; ++tt) {
            #pragma unroll
            for (int r = 0; r < 8; ++r)
                acc[tt][r] = ESN_TANH(acc[tt][r]);
            float* p = outBase + (size_t)t * 64 + tt * 16 + 8 * h;
            v4f lo = { acc[tt][0], acc[tt][1], acc[tt][2], acc[tt][3] };
            v4f hi = { acc[tt][4], acc[tt][5], acc[tt][6], acc[tt][7] };
            __builtin_nontemporal_store(lo, (v4f*)p);
            __builtin_nontemporal_store(hi, (v4f*)(p + 4));
        }

        // ---- rebuild state B fragments from D tiles (in-register half swap) ----
        // B frag f, elem e: value Y[u = f*32 + h*16 + e][m].
        // Source D tile = 2f + h; e<8 lives in half 0, e>=8 in half 1.
        #pragma unroll
        for (int f = 0; f < 2; ++f) {
            float own[8], foreign[8];
            #pragma unroll
            for (int r = 0; r < 8; ++r) {
                float a0 = acc[2 * f][r];       // tile 2f   (used by half 0)
                float a1 = acc[2 * f + 1][r];   // tile 2f+1 (used by half 1)
                own[r]      = h ? a1 : a0;      // this lane's half of its tile
                float offer = h ? a0 : a1;      // what the other half needs
                foreign[r]  = swap_half(offer); // other half of this lane's tile
            }
            v16bf nb;
            #pragma unroll
            for (int e = 0; e < 16; ++e) {
                int r = e & 7;
                float v = (((e >= 8) ? 1 : 0) == h) ? own[r] : foreign[r];
                nb[e] = (__bf16)v;
            }
            Bst[f] = nb;
        }

        curA = nxtA;
        curB = nxtB;
    }
}

extern "C" void kernel_launch(void* const* d_in, const int* in_sizes, int n_in,
                              void* d_out, int out_size, void* d_ws, size_t ws_size,
                              hipStream_t stream) {
    const float* inputs = (const float*)d_in[0];  // [B, T, 8]
    const float* W_in   = (const float*)d_in[1];  // [8, 64]
    const float* b      = (const float*)d_in[2];  // [64]
    const float* W_rec  = (const float*)d_in[3];  // [64, 64]
    float* out = (float*)d_out;                   // [B, T, 64]

    const int D = 8;
    const int B = 256;                 // per reference setup
    const int T = in_sizes[0] / (B * D);

    dim3 grid(B / 16), block(32);      // one wave per 16-row batch strip
    esn_reservoir_kernel<<<grid, block, 0, stream>>>(inputs, W_in, b, W_rec, out, T);
}
